// META3D_50843822850091
// MI455X (gfx1250) — compile-verified, matrix operand order checked by reading, not run on hardware
//
#include <hip/hip_runtime.h>
#include <math.h>

typedef _Float16 f16;
typedef __attribute__((ext_vector_type(16))) _Float16 v16h;
typedef __attribute__((ext_vector_type(8)))  _Float16 v8h;
typedef __attribute__((ext_vector_type(8)))  float    v8f;

#define NTOK 8192
#define EPS  1e-5f

// ---------- index permutations (local patch order <-> global token order) ----------
__device__ __forceinline__ int g_of_l(int nl) {
    int r = nl >> 5, p = nl & 31;
    int rd = r >> 6, rh = (r >> 3) & 7, rw = r & 7;
    int pd = p >> 4, ph = (p >> 2) & 3, pw = p & 3;
    return ((rd * 2 + pd) << 10) + ((rh * 4 + ph) << 5) + (rw * 4 + pw);
}
__device__ __forceinline__ int l_of_g(int n) {
    int d = n >> 10, h = (n >> 5) & 31, w = n & 31;
    int r = (d >> 1) * 64 + (h >> 2) * 8 + (w >> 2);
    int p = (d & 1) * 16 + (h & 3) * 4 + (w & 3);
    return r * 32 + p;
}

// ---------- WMMA fragment helpers ----------
// 16-bit operand layout (ISA 7.12.2): lane L: row/col = L&15, Kbase = (L>>4)*8;
// v16h elements 0..7 = K [Kbase..Kbase+7], elements 8..15 = K [Kbase+16..Kbase+23].
__device__ __forceinline__ v16h ldfrag(const f16* p) {
    union { v16h v; v8h h[2]; } u;
    u.h[0] = *(const v8h*)p;
    u.h[1] = *(const v8h*)(p + 16);
    return u.v;
}

// Register-blocked core: one wave computes a 16(M) x 64(N) strip.
// A-frag loaded once per k-step and reused against 4 B-frags (2.5 b128 loads / wmma).
__device__ __forceinline__ void wmma_strip4(const f16* A0, const f16* W0, int K, v8f acc[4]) {
    int lane = threadIdx.x & 31;
    int hf = lane >> 4, r = lane & 15;
    const f16* pa = A0 + (long)r * K + hf * 8;
    const f16* pb = W0 + (long)r * K + hf * 8;
    for (int kk = 0; kk < K; kk += 32) {
        v16h a = ldfrag(pa + kk);
#pragma unroll
        for (int t = 0; t < 4; t++) {
            v16h b = ldfrag(pb + (long)t * 16 * K + kk);
            acc[t] = __builtin_amdgcn_wmma_f32_16x16x32_f16(false, a, false, b,
                                                            (short)0, acc[t], false, false);
        }
    }
}

// ---------- elementwise f32 -> f16 ----------
__global__ void k_f32tof16(const float* __restrict__ s, f16* __restrict__ d, int n) {
    int i = blockIdx.x * 256 + threadIdx.x;
    if (i < n) d[i] = (f16)s[i];
}

// ---------- transpose x (C,N)->(N,C) + per-token LN with two param sets ----------
__global__ void k_token_ln(const float* __restrict__ x,
                           const float* __restrict__ lg, const float* __restrict__ lb,
                           const float* __restrict__ gg, const float* __restrict__ gb,
                           float* __restrict__ Xt, f16* __restrict__ Al, f16* __restrict__ Ag) {
    __shared__ float tile[128][33];
    __shared__ float sm[32], sr[32];
    int tid = threadIdx.x, ty = tid >> 5, lane = tid & 31;
    int n0 = blockIdx.x * 32;
    for (int c = ty; c < 128; c += 4) tile[c][lane] = x[c * NTOK + n0 + lane];
    __syncthreads();
    if (tid < 32) {
        float s = 0.f, s2 = 0.f;
        for (int c = 0; c < 128; c++) { float v = tile[c][tid]; s += v; s2 += v * v; }
        float m = s * (1.0f / 128.0f);
        float var = s2 * (1.0f / 128.0f) - m * m;
        sm[tid] = m; sr[tid] = rsqrtf(var + EPS);
    }
    __syncthreads();
    for (int j = 0; j < 8; j++) {
        int t = ty * 8 + j;
        float m = sm[t], rs = sr[t];
        int n = n0 + t, nl = l_of_g(n);
        for (int q = 0; q < 4; q++) {
            int c = lane + 32 * q;
            float v = tile[c][t];
            float z = (v - m) * rs;
            Xt[(long)n * 128 + c]  = v;
            Ag[(long)n * 128 + c]  = (f16)(z * gg[c] + gb[c]);
            Al[(long)nl * 128 + c] = (f16)(z * lg[c] + lb[c]);
        }
    }
}

// ---------- GEMM variants: block = 128 threads = 4 waves; wave = 16x64 strip ----------
// grid = (M/64, Nn/64); wave w handles rows m0 = bx*64 + w*16, cols [by*64, by*64+64)
__global__ void k_gemm_f16(const f16* __restrict__ A, const f16* __restrict__ W,
                           f16* __restrict__ Out, int Nn, int K) {
    int wid = threadIdx.x >> 5;
    int m0 = blockIdx.x * 64 + wid * 16;
    int n0 = blockIdx.y * 64;
    v8f z = {};
    v8f acc[4] = {z, z, z, z};
    wmma_strip4(A + (long)m0 * K, W + (long)n0 * K, K, acc);
    int lane = threadIdx.x & 31, hf = lane >> 4, r15 = lane & 15;
#pragma unroll
    for (int t = 0; t < 4; t++)
#pragma unroll
        for (int e = 0; e < 8; e++) {
            int row = m0 + e + 8 * hf;
            Out[(long)row * Nn + n0 + t * 16 + r15] = (f16)acc[t][e];
        }
}

__global__ void k_gemm_f32(const f16* __restrict__ A, const f16* __restrict__ W,
                           float* __restrict__ Out, int Nn, int K) {
    int wid = threadIdx.x >> 5;
    int m0 = blockIdx.x * 64 + wid * 16;
    int n0 = blockIdx.y * 64;
    v8f z = {};
    v8f acc[4] = {z, z, z, z};
    wmma_strip4(A + (long)m0 * K, W + (long)n0 * K, K, acc);
    int lane = threadIdx.x & 31, hf = lane >> 4, r15 = lane & 15;
#pragma unroll
    for (int t = 0; t < 4; t++)
#pragma unroll
        for (int e = 0; e < 8; e++) {
            int row = m0 + e + 8 * hf;
            Out[(long)row * Nn + n0 + t * 16 + r15] = acc[t][e];
        }
}

__global__ void k_gemm_bias_res(const f16* __restrict__ A, const f16* __restrict__ W,
                                const float* __restrict__ bias, const float* __restrict__ res,
                                float* __restrict__ Out, int Nn, int K, int perm) {
    int wid = threadIdx.x >> 5;
    int m0 = blockIdx.x * 64 + wid * 16;
    int n0 = blockIdx.y * 64;
    v8f z = {};
    v8f acc[4] = {z, z, z, z};
    wmma_strip4(A + (long)m0 * K, W + (long)n0 * K, K, acc);
    int lane = threadIdx.x & 31, hf = lane >> 4, r15 = lane & 15;
#pragma unroll
    for (int e = 0; e < 8; e++) {
        int row = m0 + e + 8 * hf;
        int rrow = perm ? g_of_l(row) : row;
#pragma unroll
        for (int t = 0; t < 4; t++) {
            int c = n0 + t * 16 + r15;
            Out[(long)row * Nn + c] = acc[t][e] + bias[c] + res[(long)rrow * Nn + c];
        }
    }
}

__global__ void k_gemm_bias_gelu(const f16* __restrict__ A, const f16* __restrict__ W,
                                 const float* __restrict__ bias, f16* __restrict__ Out,
                                 int Nn, int K) {
    int wid = threadIdx.x >> 5;
    int m0 = blockIdx.x * 64 + wid * 16;
    int n0 = blockIdx.y * 64;
    v8f z = {};
    v8f acc[4] = {z, z, z, z};
    wmma_strip4(A + (long)m0 * K, W + (long)n0 * K, K, acc);
    int lane = threadIdx.x & 31, hf = lane >> 4, r15 = lane & 15;
#pragma unroll
    for (int t = 0; t < 4; t++)
#pragma unroll
        for (int e = 0; e < 8; e++) {
            int row = m0 + e + 8 * hf;
            int c = n0 + t * 16 + r15;
            float v = acc[t][e] + bias[c];
            float g = 0.5f * v * (1.0f + erff(v * 0.70710678118654752f));
            Out[(long)row * Nn + c] = (f16)g;
        }
}

// ---------- rows LN (f32 in, f16 out), wave per row ----------
__global__ void k_rows_ln(const float* __restrict__ In, const float* __restrict__ g,
                          const float* __restrict__ b, f16* __restrict__ Out) {
    int wid = threadIdx.x >> 5, lane = threadIdx.x & 31;
    int row = blockIdx.x * 4 + wid;
    const float* p = In + (long)row * 128;
    float v[4]; float s = 0.f, s2 = 0.f;
#pragma unroll
    for (int q = 0; q < 4; q++) { v[q] = p[lane + 32 * q]; s += v[q]; s2 += v[q] * v[q]; }
#pragma unroll
    for (int m = 16; m >= 1; m >>= 1) { s += __shfl_xor(s, m, 32); s2 += __shfl_xor(s2, m, 32); }
    float mean = s * (1.0f / 128.0f);
    float rs = rsqrtf(s2 * (1.0f / 128.0f) - mean * mean + EPS);
#pragma unroll
    for (int q = 0; q < 4; q++) {
        int c = lane + 32 * q;
        Out[(long)row * 128 + c] = (f16)((v[q] - mean) * rs * g[c] + b[c]);
    }
}

// ---------- local windowed attention (patch=32 tokens, 8 heads, head-dim 16) ----------
__global__ void k_local_attn(const f16* __restrict__ qkv, f16* __restrict__ ol) {
    __shared__ float L[32 * 384];
    int r = blockIdx.x;
    int tid = threadIdx.x;
    for (int e = tid; e < 32 * 384; e += 256) L[e] = (float)qkv[(long)r * 32 * 384 + e];
    __syncthreads();
    int h = tid >> 5, n = tid & 31;
    const float* qp = &L[n * 384 + h * 16];
    float s[32];
    float mx = -1e30f;
#pragma unroll
    for (int m2 = 0; m2 < 32; m2++) {
        const float* kp = &L[m2 * 384 + 128 + h * 16];
        float acc = 0.f;
#pragma unroll
        for (int d = 0; d < 16; d++) acc += qp[d] * kp[d];
        acc *= 0.25f;
        s[m2] = acc; mx = fmaxf(mx, acc);
    }
    float den = 0.f;
#pragma unroll
    for (int m2 = 0; m2 < 32; m2++) { s[m2] = expf(s[m2] - mx); den += s[m2]; }
    float inv = 1.0f / den;
#pragma unroll
    for (int d = 0; d < 16; d++) {
        float o = 0.f;
#pragma unroll
        for (int m2 = 0; m2 < 32; m2++) o += s[m2] * L[m2 * 384 + 256 + h * 16 + d];
        o *= inv;
        // bug-faithful torch layout: token' = h*4 + d/4, chan' = (d%4)*32 + n
        int np = h * 4 + (d >> 2);
        int cp = (d & 3) * 32 + n;
        ol[((long)r * 32 + np) * 128 + cp] = (f16)o;
    }
}

// ---------- global branch helpers ----------
__global__ void k_build_qpad(const f16* __restrict__ qkv, f16* __restrict__ Qg) {
    int idx = blockIdx.x * 256 + threadIdx.x;
    if (idx >= NTOK * 128) return;
    int n = idx >> 7, c = idx & 127;
    int h = c >> 4, d = c & 15;
    long o = ((long)h * NTOK + n) * 32;
    Qg[o + d]      = qkv[(long)n * 384 + c];
    Qg[o + 16 + d] = (f16)0.0f;
}

__global__ void k_build_cols(const f16* __restrict__ qkv, f16* __restrict__ Kc, f16* __restrict__ Vc) {
    int idx = blockIdx.x * 256 + threadIdx.x;
    if (idx >= 1024 * 1024) return;
    int m = idx >> 10, kp = idx & 1023;
    int ci = kp >> 3, ki = kp & 7;
    int kd = ki >> 2, kh = (ki >> 1) & 1, kw = ki & 1;
    int od = m >> 8, oh = (m >> 4) & 15, ow = m & 15;
    int tok = (2 * od + kd) * 1024 + (2 * oh + kh) * 32 + (2 * ow + kw);
    Kc[idx] = qkv[(long)tok * 384 + 128 + ci];
    Vc[idx] = qkv[(long)tok * 384 + 256 + ci];
}

__global__ void k_kv_ln_pack(const float* __restrict__ Kr, const float* __restrict__ Vr,
                             const float* __restrict__ nkg, const float* __restrict__ nkb,
                             const float* __restrict__ nvg, const float* __restrict__ nvb,
                             f16* __restrict__ Krp, f16* __restrict__ VrT) {
    int idx = blockIdx.x * 256 + threadIdx.x;
    if (idx >= 1024 * 8) return;
    int m = idx >> 3, h = idx & 7;
    float kv[16], vv[16];
    float ks = 0.f, ks2 = 0.f, vs = 0.f, vs2 = 0.f;
#pragma unroll
    for (int d = 0; d < 16; d++) {
        kv[d] = Kr[m * 128 + h * 16 + d]; ks += kv[d]; ks2 += kv[d] * kv[d];
        vv[d] = Vr[m * 128 + h * 16 + d]; vs += vv[d]; vs2 += vv[d] * vv[d];
    }
    float km = ks * (1.0f / 16.0f), krs = rsqrtf(ks2 * (1.0f / 16.0f) - km * km + EPS);
    float vm = vs * (1.0f / 16.0f), vrs = rsqrtf(vs2 * (1.0f / 16.0f) - vm * vm + EPS);
    long ko = ((long)h * 1024 + m) * 32;
#pragma unroll
    for (int d = 0; d < 16; d++) {
        Krp[ko + d]      = (f16)((kv[d] - km) * krs * nkg[d] + nkb[d]);
        Krp[ko + 16 + d] = (f16)0.0f;
        VrT[(long)h * 16384 + d * 1024 + m] = (f16)((vv[d] - vm) * vrs * nvg[d] + nvb[d]);
    }
}

// ---------- global flash attention: Q(8192x16) vs Kr/Vr(1024x16) per head ----------
__global__ void k_flash(const f16* __restrict__ Qg, const f16* __restrict__ Krp,
                        const f16* __restrict__ VrT, f16* __restrict__ og) {
    __shared__ f16 Pl[4][16 * 32];
    int wid = threadIdx.x >> 5, lane = threadIdx.x & 31;
    int hf = lane >> 4, r15 = lane & 15;
    int h = blockIdx.y;
    int q0 = (blockIdx.x * 4 + wid) * 16;
    const f16* Qbase = Qg + ((long)h * NTOK + q0) * 32;
    const f16* Kbase = Krp + (long)h * 1024 * 32;
    const f16* Vbase = VrT + (long)h * 16 * 1024;
    v16h qf = ldfrag(Qbase + r15 * 32 + hf * 8);
    float mrow[8], lrow[8];
    v8f oacc = {};
#pragma unroll
    for (int e = 0; e < 8; e++) { mrow[e] = -1e30f; lrow[e] = 0.0f; }
    f16* P = &Pl[wid][0];
    for (int kt = 0; kt < 1024; kt += 32) {
        v16h kf0 = ldfrag(Kbase + (long)(kt + r15) * 32 + hf * 8);
        v16h kf1 = ldfrag(Kbase + (long)(kt + 16 + r15) * 32 + hf * 8);
        v8f s0 = {}, s1 = {};
        s0 = __builtin_amdgcn_wmma_f32_16x16x32_f16(false, qf, false, kf0, (short)0, s0, false, false);
        s1 = __builtin_amdgcn_wmma_f32_16x16x32_f16(false, qf, false, kf1, (short)0, s1, false, false);
        float tm[8], ts[8], al[8];
#pragma unroll
        for (int e = 0; e < 8; e++) {
            s0[e] *= 0.25f; s1[e] *= 0.25f;
            tm[e] = fmaxf(s0[e], s1[e]);
        }
#pragma unroll
        for (int e = 0; e < 8; e++)
            for (int msk = 8; msk >= 1; msk >>= 1)
                tm[e] = fmaxf(tm[e], __shfl_xor(tm[e], msk, 32));
#pragma unroll
        for (int e = 0; e < 8; e++) {
            float nm = fmaxf(mrow[e], tm[e]);
            al[e] = expf(mrow[e] - nm);
            mrow[e] = nm;
            float p0 = expf(s0[e] - nm), p1 = expf(s1[e] - nm);
            s0[e] = p0; s1[e] = p1;
            ts[e] = p0 + p1;
        }
#pragma unroll
        for (int e = 0; e < 8; e++)
            for (int msk = 8; msk >= 1; msk >>= 1)
                ts[e] += __shfl_xor(ts[e], msk, 32);
#pragma unroll
        for (int e = 0; e < 8; e++) {
            lrow[e] = lrow[e] * al[e] + ts[e];
            oacc[e] *= al[e];
        }
        __syncthreads();
#pragma unroll
        for (int e = 0; e < 8; e++) {
            int row = e + 8 * hf;                    // C-frag row mapping
            P[row * 32 + r15]      = (f16)s0[e];
            P[row * 32 + 16 + r15] = (f16)s1[e];
        }
        __syncthreads();
        v16h pf = ldfrag(P + r15 * 32 + hf * 8);     // A-frag of P (16x32)
        v16h vf = ldfrag(Vbase + (long)r15 * 1024 + kt + hf * 8); // B-frag of V (32x16)
        oacc = __builtin_amdgcn_wmma_f32_16x16x32_f16(false, pf, false, vf, (short)0, oacc, false, false);
    }
#pragma unroll
    for (int e = 0; e < 8; e++) {
        int row = q0 + e + 8 * hf;
        og[(long)row * 128 + h * 16 + r15] = (f16)(oacc[e] / lrow[e]);
    }
}

// ---------- final gate ----------
__global__ void k_gate(const float* __restrict__ x, const float* __restrict__ Lf,
                       const float* __restrict__ Gf, float* __restrict__ out) {
    int idx = blockIdx.x * 256 + threadIdx.x;
    if (idx >= 128 * NTOK) return;
    int n = idx & (NTOK - 1);
    int c = idx >> 13;
    int nl = l_of_g(n);
    float s = Lf[(long)nl * 128 + c] + Gf[(long)n * 128 + c];
    out[idx] = x[idx] * (1.0f / (1.0f + expf(-s)));
}

// =======================================================================================
extern "C" void kernel_launch(void* const* d_in, const int* in_sizes, int n_in,
                              void* d_out, int out_size, void* d_ws, size_t ws_size,
                              hipStream_t stream) {
    const float* x       = (const float*)d_in[0];
    const float* l_n1_g  = (const float*)d_in[1];
    const float* l_n1_b  = (const float*)d_in[2];
    const float* l_qkv_w = (const float*)d_in[3];
    const float* l_proj_w= (const float*)d_in[4];
    const float* l_proj_b= (const float*)d_in[5];
    const float* l_n2_g  = (const float*)d_in[6];
    const float* l_n2_b  = (const float*)d_in[7];
    const float* l_fc1_w = (const float*)d_in[8];
    const float* l_fc1_b = (const float*)d_in[9];
    const float* l_fc2_w = (const float*)d_in[10];
    const float* l_fc2_b = (const float*)d_in[11];
    const float* g_n1_g  = (const float*)d_in[12];
    const float* g_n1_b  = (const float*)d_in[13];
    const float* g_qkv_w = (const float*)d_in[14];
    const float* g_proj_w= (const float*)d_in[15];
    const float* g_proj_b= (const float*)d_in[16];
    const float* g_n2_g  = (const float*)d_in[17];
    const float* g_n2_b  = (const float*)d_in[18];
    const float* g_fc1_w = (const float*)d_in[19];
    const float* g_fc1_b = (const float*)d_in[20];
    const float* g_fc2_w = (const float*)d_in[21];
    const float* g_fc2_b = (const float*)d_in[22];
    const float* g_ke_w  = (const float*)d_in[23];
    const float* g_ve_w  = (const float*)d_in[24];
    const float* g_nk_g  = (const float*)d_in[25];
    const float* g_nk_b  = (const float*)d_in[26];
    const float* g_nv_g  = (const float*)d_in[27];
    const float* g_nv_b  = (const float*)d_in[28];
    (void)in_sizes; (void)n_in; (void)out_size; (void)ws_size;

    char* ws = (char*)d_ws;
    size_t off = 0;
    auto alloc = [&](size_t bytes) -> char* {
        char* p = ws + off;
        off += (bytes + 255) & ~(size_t)255;
        return p;
    };
    float* Xt    = (float*)alloc((size_t)NTOK * 128 * 4);
    f16*   Al    = (f16*)  alloc((size_t)NTOK * 128 * 2);
    f16*   Ag    = (f16*)  alloc((size_t)NTOK * 128 * 2);
    f16*   QKVl  = (f16*)  alloc((size_t)NTOK * 384 * 2);
    f16*   QKVg  = (f16*)  alloc((size_t)NTOK * 384 * 2);
    f16*   Ol    = (f16*)  alloc((size_t)NTOK * 128 * 2);
    float* Lres  = (float*)alloc((size_t)NTOK * 128 * 4);
    f16*   LN2   = (f16*)  alloc((size_t)NTOK * 128 * 2);
    f16*   Hbuf  = (f16*)  alloc((size_t)NTOK * 512 * 2);
    float* Lfin  = (float*)alloc((size_t)NTOK * 128 * 4);
    f16*   Qg    = (f16*)  alloc((size_t)8 * NTOK * 32 * 2);
    f16*   Kcol  = (f16*)  alloc((size_t)1024 * 1024 * 2);
    f16*   Vcol  = (f16*)  alloc((size_t)1024 * 1024 * 2);
    float* Krr   = (float*)alloc((size_t)1024 * 128 * 4);
    float* Vrr   = (float*)alloc((size_t)1024 * 128 * 4);
    f16*   Krp   = (f16*)  alloc((size_t)8 * 1024 * 32 * 2);
    f16*   VrT   = (f16*)  alloc((size_t)8 * 16 * 1024 * 2);
    f16*   Og    = (f16*)  alloc((size_t)NTOK * 128 * 2);
    float* Gres  = (float*)alloc((size_t)NTOK * 128 * 4);
    float* Gfin  = (float*)alloc((size_t)NTOK * 128 * 4);
    f16* Wqkv_l = (f16*)alloc(49152 * 2);
    f16* Wproj_l= (f16*)alloc(16384 * 2);
    f16* Wfc1_l = (f16*)alloc(65536 * 2);
    f16* Wfc2_l = (f16*)alloc(65536 * 2);
    f16* Wqkv_g = (f16*)alloc(49152 * 2);
    f16* Wproj_g= (f16*)alloc(16384 * 2);
    f16* Wfc1_g = (f16*)alloc(65536 * 2);
    f16* Wfc2_g = (f16*)alloc(65536 * 2);
    f16* Wke    = (f16*)alloc(131072 * 2);
    f16* Wve    = (f16*)alloc(131072 * 2);

    // weight conversion f32 -> f16
#define CVT(src, dst, n) k_f32tof16<<<((n) + 255) / 256, 256, 0, stream>>>(src, dst, n)
    CVT(l_qkv_w, Wqkv_l, 49152);  CVT(l_proj_w, Wproj_l, 16384);
    CVT(l_fc1_w, Wfc1_l, 65536);  CVT(l_fc2_w, Wfc2_l, 65536);
    CVT(g_qkv_w, Wqkv_g, 49152);  CVT(g_proj_w, Wproj_g, 16384);
    CVT(g_fc1_w, Wfc1_g, 65536);  CVT(g_fc2_w, Wfc2_g, 65536);
    CVT(g_ke_w,  Wke,   131072);  CVT(g_ve_w,  Wve,   131072);
#undef CVT

    // transpose + LN1 (both branches, both token orders)
    k_token_ln<<<NTOK / 32, 128, 0, stream>>>(x, l_n1_g, l_n1_b, g_n1_g, g_n1_b, Xt, Al, Ag);

    // QKV GEMMs (8192 x 384 x 128)
    k_gemm_f16<<<dim3(NTOK / 64, 6), 128, 0, stream>>>(Al, Wqkv_l, QKVl, 384, 128);
    k_gemm_f16<<<dim3(NTOK / 64, 6), 128, 0, stream>>>(Ag, Wqkv_g, QKVg, 384, 128);

    // ---- local branch ----
    k_local_attn<<<256, 256, 0, stream>>>(QKVl, Ol);
    k_gemm_bias_res<<<dim3(NTOK / 64, 2), 128, 0, stream>>>(Ol, Wproj_l, l_proj_b, Xt, Lres, 128, 128, 1);
    k_rows_ln<<<NTOK / 4, 128, 0, stream>>>(Lres, l_n2_g, l_n2_b, LN2);
    k_gemm_bias_gelu<<<dim3(NTOK / 64, 8), 128, 0, stream>>>(LN2, Wfc1_l, l_fc1_b, Hbuf, 512, 128);
    k_gemm_bias_res<<<dim3(NTOK / 64, 2), 128, 0, stream>>>(Hbuf, Wfc2_l, l_fc2_b, Lres, Lfin, 128, 512, 0);

    // ---- global branch ----
    k_build_qpad<<<(NTOK * 128) / 256, 256, 0, stream>>>(QKVg, Qg);
    k_build_cols<<<(1024 * 1024) / 256, 256, 0, stream>>>(QKVg, Kcol, Vcol);
    k_gemm_f32<<<dim3(1024 / 64, 2), 128, 0, stream>>>(Kcol, Wke, Krr, 128, 1024);
    k_gemm_f32<<<dim3(1024 / 64, 2), 128, 0, stream>>>(Vcol, Wve, Vrr, 128, 1024);
    k_kv_ln_pack<<<(1024 * 8) / 256, 256, 0, stream>>>(Krr, Vrr, g_nk_g, g_nk_b, g_nv_g, g_nv_b, Krp, VrT);
    k_flash<<<dim3(NTOK / 64, 8), 128, 0, stream>>>(Qg, Krp, VrT, Og);
    k_gemm_bias_res<<<dim3(NTOK / 64, 2), 128, 0, stream>>>(Og, Wproj_g, g_proj_b, Xt, Gres, 128, 128, 0);
    k_rows_ln<<<NTOK / 4, 128, 0, stream>>>(Gres, g_n2_g, g_n2_b, LN2);
    k_gemm_bias_gelu<<<dim3(NTOK / 64, 8), 128, 0, stream>>>(LN2, Wfc1_g, g_fc1_b, Hbuf, 512, 128);
    k_gemm_bias_res<<<dim3(NTOK / 64, 2), 128, 0, stream>>>(Hbuf, Wfc2_g, g_fc2_b, Gres, Gfin, 128, 512, 0);

    // ---- gate ----
    k_gate<<<(128 * NTOK) / 256, 256, 0, stream>>>(x, Lfin, Gfin, (float*)d_out);
}
